// TreeLSTM_35021163331693
// MI455X (gfx1250) — compile-verified, matrix operand order checked by reading, not run on hardware
//
#include <hip/hip_runtime.h>
#include <hip/hip_bf16.h>
#include <cstddef>

// ---------------------------------------------------------------------------
// TreeLSTM level sweep on MI455X (gfx1250), bf16 WMMA path, TM=32 M-tiles.
//   B=16, N=32767 (depth 15), D=H=128.
//   d_out = [ h (B*N*H f32) | c (B*N*H f32) ]  -- also used as h/c storage.
//   d_ws  = fused transposed bf16 weight matrix Wt[768][512].
// ---------------------------------------------------------------------------

typedef __attribute__((ext_vector_type(16))) __bf16    v16bf;
typedef __attribute__((ext_vector_type(8)))  float     v8f;
typedef __attribute__((ext_vector_type(4)))  unsigned  u32x4;

union Frag {
    v16bf v;
    struct { u32x4 lo, hi; } u;
};

__device__ __forceinline__ unsigned short f2bf(float f) {
    union { float f; unsigned u; } x; x.f = f;
    unsigned u = x.u;
    unsigned r = u + 0x7FFFu + ((u >> 16) & 1u);   // round-to-nearest-even
    return (unsigned short)(r >> 16);
}

__device__ __forceinline__ float sigmoidf_fast(float x) {
    return 1.0f / (1.0f + __expf(-x));
}

// ---------------------------------------------------------------------------
// Pack fused transposed weight matrix Wt (row-major [n=768][k=512] bf16):
//   n in [0,384)   : k<128 -> W_iou[k][n] ; 128<=k<256 -> U_iou[k-128][n]
//   n in [384,512) : k<128 -> W_f[k][n-384]
//   n in [512,640) : 256<=k<384 -> U_f[k-256][n-512]    (child 0)
//   n in [640,768) : 384<=k<512 -> U_f[k-384][n-640]    (child 1)
// ---------------------------------------------------------------------------
__global__ __launch_bounds__(256) void pack_weights_kernel(
    const float* __restrict__ W_iou, const float* __restrict__ U_iou,
    const float* __restrict__ W_f,   const float* __restrict__ U_f,
    unsigned short* __restrict__ Wt)
{
    int idx = blockIdx.x * 256 + threadIdx.x;
    if (idx >= 768 * 512) return;
    int n = idx >> 9;
    int k = idx & 511;
    float v = 0.0f;
    if (n < 384) {
        if (k < 128)       v = W_iou[k * 384 + n];
        else if (k < 256)  v = U_iou[(k - 128) * 384 + n];
    } else if (n < 512) {
        if (k < 128)       v = W_f[k * 128 + (n - 384)];
    } else if (n < 640) {
        if (k >= 256 && k < 384) v = U_f[(k - 256) * 128 + (n - 512)];
    } else {
        if (k >= 384)      v = U_f[(k - 384) * 128 + (n - 640)];
    }
    Wt[idx] = f2bf(v);
}

// ---------------------------------------------------------------------------
// One tree level. 256 threads = 8 wave32. 32-row M tile per workgroup
// (two 16-row WMMA subtiles sharing each B fragment -> 2x B reuse).
// Row m = b * npar + pl ; node p = (2^lvl - 1) + pl.  Rows clamp to the last
// valid row when M_total < 32 (level 0) -- redundant identical work, EXEC
// stays all-ones for WMMA.
// A_ext row (512 bf16): [ x(128) | h0+h1(128) | h0(128) | h1(128) ]
// Output columns (768 f32): [ iou(384) | fpar(128) | fc0(128) | fc1(128) ]
// ---------------------------------------------------------------------------
template <bool INTERNAL>
__global__ __launch_bounds__(256) void tree_level_kernel(
    const float* __restrict__ features,
    const float* __restrict__ b_iou,
    const float* __restrict__ b_f,
    const unsigned short* __restrict__ Wt,
    float* __restrict__ hbuf,
    float* __restrict__ cbuf,
    int lvl)
{
    constexpr int Nn = 32767, Dd = 128, Hh = 128;
    constexpr int AST = 520;   // A row stride (halfs): 512 + 8 pad -> conflict-free
    constexpr int OST = 776;   // O row stride (floats): 768 + 8 pad

    __shared__ __align__(16) unsigned short A[32 * AST];   // ~33 KB
    __shared__ float O[32 * OST];                          // ~99 KB

    const int t     = threadIdx.x;
    const int m0    = blockIdx.x * 32;
    const int npm1  = (1 << lvl) - 1;       // npar - 1
    const int sbase = npm1;                 // first node index of level
    const int mMax  = (16 << lvl) - 1;      // last valid row (B * npar - 1)

    // ---------------- Phase 1: gather A tile into LDS (f32 -> bf16) --------
    {
        const int r   = t >> 3;             // row 0..31
        const int c16 = (t & 7) * 16;       // 16 columns per thread per segment
        int m = m0 + r; if (m > mMax) m = mMax;
        const int b = m >> lvl;
        const int p = sbase + (m & npm1);
        const float* xr = features + ((size_t)b * Nn + p) * Dd + c16;
        unsigned short* Ar = &A[r * AST];
        #pragma unroll
        for (int j = 0; j < 16; ++j) Ar[c16 + j] = f2bf(xr[j]);
        if (INTERNAL) {
            const float* h0 = hbuf + ((size_t)b * Nn + (2 * p + 1)) * Hh + c16;
            const float* h1 = h0 + Hh;
            #pragma unroll
            for (int j = 0; j < 16; ++j) {
                float a0 = h0[j], a1 = h1[j];
                Ar[128 + c16 + j] = f2bf(a0 + a1);
                Ar[256 + c16 + j] = f2bf(a0);
                Ar[384 + c16 + j] = f2bf(a1);
            }
        }
    }
    __syncthreads();

    // ---------------- Phase 2: WMMA bf16 GEMM, 2 M-subtiles per B frag -----
    {
        const int wave = t >> 5;
        const int lane = t & 31;
        const int nsel = lane & 15;           // N column / A row within tile
        const int koff = (lane >> 4) * 8;     // K sub-offset per half-wave
        const int ntiles = INTERNAL ? 6 : 3;  // 48 (or 24) N-tiles over 8 waves

        for (int i = 0; i < ntiles; ++i) {
            const int nt = wave + i * 8;      // N-tile id 0..47
            int kstart, kcount;
            if (nt < 24)      { kstart = 0;   kcount = INTERNAL ? 256 : 128; } // iou
            else if (nt < 32) { kstart = 0;   kcount = 128; }                  // fpar
            else if (nt < 40) { kstart = 256; kcount = 128; }                  // fc0
            else              { kstart = 384; kcount = 128; }                  // fc1

            v8f acc0 = {0.f, 0.f, 0.f, 0.f, 0.f, 0.f, 0.f, 0.f};
            v8f acc1 = {0.f, 0.f, 0.f, 0.f, 0.f, 0.f, 0.f, 0.f};
            const unsigned short* brow  = Wt + (size_t)(nt * 16 + nsel) * 512;
            const unsigned short* arow0 = &A[nsel * AST];          // rows 0..15
            const unsigned short* arow1 = &A[(16 + nsel) * AST];   // rows 16..31

            for (int kc = 0; kc < kcount; kc += 32) {
                const int k0 = kstart + kc;
                Frag bf, a0, a1;
                bf.u.lo = *(const u32x4*)(brow + k0 + koff);
                bf.u.hi = *(const u32x4*)(brow + k0 + 16 + koff);
                a0.u.lo = *(const u32x4*)(arow0 + k0 + koff);
                a0.u.hi = *(const u32x4*)(arow0 + k0 + 16 + koff);
                a1.u.lo = *(const u32x4*)(arow1 + k0 + koff);
                a1.u.hi = *(const u32x4*)(arow1 + k0 + 16 + koff);
                acc0 = __builtin_amdgcn_wmma_f32_16x16x32_bf16(
                        false, a0.v, false, bf.v, (short)0, acc0, false, false);
                acc1 = __builtin_amdgcn_wmma_f32_16x16x32_bf16(
                        false, a1.v, false, bf.v, (short)0, acc1, false, false);
            }
            // D layout: lane holds column nsel; VGPR r -> row r + 8*(lane>=16)
            const int mb = (lane >> 4) * 8;
            #pragma unroll
            for (int vr = 0; vr < 8; ++vr) {
                O[(mb + vr) * OST + nt * 16 + nsel]        = acc0[vr];
                O[(16 + mb + vr) * OST + nt * 16 + nsel]   = acc1[vr];
            }
        }
    }
    __syncthreads();

    // ---------------- Phase 3: gates + combine + store h,c -----------------
    {
        const int r  = t >> 3;              // row 0..31
        const int j0 = (t & 7) * 16;        // 16 elements per thread
        int m = m0 + r; if (m > mMax) m = mMax;   // clamped rows rewrite same data
        const int b = m >> lvl;
        const int p = sbase + (m & npm1);
        float* crow = cbuf + ((size_t)b * Nn + p) * Hh;
        float* hrow = hbuf + ((size_t)b * Nn + p) * Hh;
        const float* Or = &O[r * OST];
        const float* cc0 = INTERNAL ? (cbuf + ((size_t)b * Nn + (2 * p + 1)) * Hh) : nullptr;
        const float* cc1 = INTERNAL ? (cbuf + ((size_t)b * Nn + (2 * p + 2)) * Hh) : nullptr;

        #pragma unroll
        for (int jj = 0; jj < 16; ++jj) {
            const int j = j0 + jj;
            float iv = sigmoidf_fast(Or[j]       + b_iou[j]);
            float ov = sigmoidf_fast(Or[128 + j] + b_iou[128 + j]);
            float uv = tanhf        (Or[256 + j] + b_iou[256 + j]);
            float cn = iv * uv;
            if (INTERNAL) {
                float fp = Or[384 + j] + b_f[j];          // NOTE: no sigmoid (faithful)
                float f0 = fp + Or[512 + j];
                float f1 = fp + Or[640 + j];
                cn += f0 * cc0[j] + f1 * cc1[j];
            }
            crow[j] = cn;
            hrow[j] = ov * tanhf(cn);
        }
    }
}

// ---------------------------------------------------------------------------
extern "C" void kernel_launch(void* const* d_in, const int* in_sizes, int n_in,
                              void* d_out, int out_size, void* d_ws, size_t ws_size,
                              hipStream_t stream)
{
    (void)in_sizes; (void)n_in; (void)out_size; (void)ws_size;
    const float* features = (const float*)d_in[0];
    const float* W_iou    = (const float*)d_in[1];
    const float* b_iou    = (const float*)d_in[2];
    const float* U_iou    = (const float*)d_in[3];
    const float* W_f      = (const float*)d_in[4];
    const float* b_f      = (const float*)d_in[5];
    const float* U_f      = (const float*)d_in[6];

    float* hbuf = (float*)d_out;
    float* cbuf = hbuf + (size_t)16 * 32767 * 128;
    unsigned short* Wt = (unsigned short*)d_ws;   // 768*512 bf16 = 768 KiB

    pack_weights_kernel<<<(768 * 512 + 255) / 256, 256, 0, stream>>>(
        W_iou, U_iou, W_f, U_f, Wt);

    // Leaves (level 14): iou only. M = 16*2^14 rows, 32 rows per block.
    tree_level_kernel<false><<<1 << 13, 256, 0, stream>>>(
        features, b_iou, b_f, Wt, hbuf, cbuf, 14);

    // Internal levels 13..0, bottom-up; stream ordering provides the sync.
    for (int l = 13; l >= 0; --l) {
        int blocks = (l == 0) ? 1 : (1 << (l - 1));
        tree_level_kernel<true><<<blocks, 256, 0, stream>>>(
            features, b_iou, b_f, Wt, hbuf, cbuf, l);
    }
}
